// SSGraphDTI_67095979098124
// MI455X (gfx1250) — matrix operand order repaired and checked by reading, not compiled, not run on hardware
//
#include <hip/hip_runtime.h>

// ---------------- problem constants (from reference) ----------------
#define BSZ   64
#define NDN   32       // drug nodes
#define NPRN  32       // protein nodes
#define LDL   100      // drug seq len / drug node feat dim
#define LPL   1000     // protein seq len / protein node feat dim
#define DIMC  64       // embed dim
#define EDD   128      // ddi edges
#define EPP   128      // ppi edges
#define EDP   64       // dpi edges
#define VDV   71
#define VPV   26

// ---------------- WMMA / TDM types ----------------
typedef __attribute__((ext_vector_type(16))) __bf16    v16bf;
typedef __attribute__((ext_vector_type(8)))  float     v8f;
typedef __attribute__((ext_vector_type(4)))  unsigned  v4u;
typedef __attribute__((ext_vector_type(8)))  int       v8i;
typedef __attribute__((ext_vector_type(4)))  int       v4i;

#if __has_builtin(__builtin_amdgcn_tensor_load_to_lds)
#define HAVE_TDM 1
#else
#define HAVE_TDM 0
#endif

__device__ __forceinline__ __bf16 f2bf(float f) {
  union { float f; unsigned u; } x; x.f = f;
  unsigned r = x.u + 0x7FFFu + ((x.u >> 16) & 1u);   // round-to-nearest-even
  unsigned short h = (unsigned short)(r >> 16);
  __bf16 b;
  __builtin_memcpy(&b, &h, sizeof(h));
  return b;
}

#if HAVE_TDM
// LDS byte offset of a __shared__ object: generic LDS addresses keep the LDS
// offset in addr[31:0] (aperture in the high half), so truncation is the offset.
__device__ __forceinline__ unsigned lds_off(const void* p) {
  return (unsigned)(unsigned long long)p;
}
// Issue one 2D-tile TDM load: tile (tile1 rows x tile0 dwords) of f32 from a
// row-major matrix with row stride `stride0` (elements) into LDS at `loff`,
// zero-filling reads past (td0, td1). pad_enable inserts 1 dword of LDS padding
// every 32 dwords -> LDS layout [rows][33] (bank-conflict-free column reads).
__device__ __forceinline__ void tdm_load_2d(const float* gptr, unsigned loff,
                                            unsigned tile0, unsigned tile1,
                                            unsigned td0, unsigned td1,
                                            unsigned stride0) {
  unsigned long long ga = (unsigned long long)gptr;
  v4u g0;
  g0[0] = 1u;                                            // count=1 (valid)
  g0[1] = loff;                                          // lds_addr
  g0[2] = (unsigned)(ga & 0xffffffffu);                  // global_addr[31:0]
  g0[3] = (unsigned)((ga >> 32) & 0x01ffffffu)           // global_addr[56:32]
        | (2u << 30);                                    // type=2 ("image")
  v8i g1;
  g1[0] = (int)((2u << 16) | (1u << 20) | (4u << 22));   // data_size=4B, pad_en, interval=32dw, pad=1dw
  g1[1] = (int)((td0 & 0xffffu) << 16);                  // tensor_dim0[15:0]
  g1[2] = (int)(((td0 >> 16) & 0xffffu) | ((td1 & 0xffffu) << 16));
  g1[3] = (int)(((td1 >> 16) & 0xffffu) | ((tile0 & 0xffffu) << 16));
  g1[4] = (int)(tile1 & 0xffffu);                        // tile_dim1 (tile_dim2=0)
  g1[5] = (int)stride0;                                  // tensor_dim0_stride[31:0]
  g1[6] = 0;
  g1[7] = 0;
  v4i z4 = {0, 0, 0, 0};
  v8i z8 = {0, 0, 0, 0, 0, 0, 0, 0};
  __builtin_amdgcn_tensor_load_to_lds(g0, g1, z4, z4, z8, 0);
}
#endif

// ---------------- generic bf16-WMMA GEMM ----------------
// C[M,N] = act( out_scale * ( A (.) B^T + bias[n] + beta*C ) )
// AMODE  0: A dense row-major (M,K)   1: implicit im2col of an NLC activation
// BTRANS 0: B row-major (N,K) (x@W.T) 1: B row-major (K,N)
struct GemmP {
  const float* A; const float* Bw; const float* bias; float* C;
  int M, N, K;
  int L, Cch, KW, Lout;          // im2col params
  unsigned mLout, mKW;           // magic reciprocals: ceil(2^32/d)
  int act;                       // 0 none, 1 relu, 2 leaky(0.01)
  float beta, out_scale;
};

template<int AMODE, int BTRANS>
__global__ __launch_bounds__(256) void gemm_bf16_wmma(GemmP p) {
  __shared__ float As[2][64][33];
  __shared__ float Bs[2][32][33];
  const int tid  = threadIdx.x;
  const int wave = tid >> 5, lane = tid & 31;
  const int wrow = wave >> 1, wcol = wave & 1;     // 4x2 waves -> 64x32 C tile
  const int half = lane >> 4, l16 = lane & 15;
  const int bm = blockIdx.y * 64, bn = blockIdx.x * 32;
  const int nK = (p.K + 31) >> 5;

#if HAVE_TDM
  auto issueB = [&](int kt) {
    const int k0 = kt << 5;
    if (BTRANS == 0)
      tdm_load_2d(p.Bw + (size_t)bn * p.K + k0, lds_off(&Bs[kt & 1][0][0]),
                  32u, 32u, (unsigned)(p.K - k0), (unsigned)(p.N - bn), (unsigned)p.K);
    else
      tdm_load_2d(p.Bw + (size_t)k0 * p.N + bn, lds_off(&Bs[kt & 1][0][0]),
                  32u, 32u, (unsigned)(p.N - bn), (unsigned)(p.K - k0), (unsigned)p.N);
  };
  auto issueA = [&](int kt) {
    const int k0 = kt << 5;
    tdm_load_2d(p.A + (size_t)bm * p.K + k0, lds_off(&As[kt & 1][0][0]),
                32u, 64u, (unsigned)(p.K - k0), (unsigned)(p.M - bm), (unsigned)p.K);
  };
  if (wave == 0) issueB(0);
  if (AMODE == 0 && wave == 1) issueA(0);
#endif

  v8f acc = {};
  for (int kt = 0; kt < nK; ++kt) {
    const int k0 = kt << 5;
#if HAVE_TDM
    const int abuf = (AMODE == 0) ? (kt & 1) : 0;
    const int bbuf = kt & 1;
#else
    const int abuf = 0, bbuf = 0;
#endif
    if (AMODE == 1) {
      // implicit im2col: A[(b*Lout+l), (ic*KW+kk)] = act[((b*L)+(l+kk))*Cch + ic]
      #pragma unroll
      for (int i = 0; i < 8; ++i) {
        int idx = i * 256 + tid;
        int m = idx >> 5, k = idx & 31;
        int gm = bm + m, gk = k0 + k;
        float v = 0.0f;
        if (gm < p.M && gk < p.K) {
          unsigned b  = (unsigned)(((unsigned long long)(unsigned)gm * p.mLout) >> 32);
          unsigned l  = (unsigned)gm - b * (unsigned)p.Lout;
          unsigned ic = (unsigned)(((unsigned long long)(unsigned)gk * p.mKW) >> 32);
          unsigned kk = (unsigned)gk - ic * (unsigned)p.KW;
          v = p.A[((size_t)(b * p.L + l + kk)) * p.Cch + ic];
        }
        As[0][m][k] = v;
      }
    }
#if HAVE_TDM
    if (wave == 0) {
      if (kt + 1 < nK) { issueB(kt + 1); __builtin_amdgcn_s_wait_tensorcnt(1); }
      else             {                 __builtin_amdgcn_s_wait_tensorcnt(0); }
    }
    if (AMODE == 0 && wave == 1) {
      if (kt + 1 < nK) { issueA(kt + 1); __builtin_amdgcn_s_wait_tensorcnt(1); }
      else             {                 __builtin_amdgcn_s_wait_tensorcnt(0); }
    }
#else
    if (AMODE == 0) {
      #pragma unroll
      for (int i = 0; i < 8; ++i) {
        int idx = i * 256 + tid;
        int m = idx >> 5, k = idx & 31;
        int gm = bm + m, gk = k0 + k;
        As[0][m][k] = (gm < p.M && gk < p.K) ? p.A[(size_t)gm * p.K + gk] : 0.0f;
      }
    }
    #pragma unroll
    for (int i = 0; i < 4; ++i) {
      int idx = i * 256 + tid;
      int n = idx >> 5, k = idx & 31;
      int gn = bn + n, gk = k0 + k;
      float v = 0.0f;
      if (gn < p.N && gk < p.K)
        v = BTRANS ? p.Bw[(size_t)gk * p.N + gn] : p.Bw[(size_t)gn * p.K + gk];
      Bs[0][n][k] = v;
    }
#endif
    __syncthreads();
    // fragments per ISA 7.12.2 layouts (bf16 conversion fused here)
    v16bf af, bf;
    const int nloc = wcol * 16 + l16;
    #pragma unroll
    for (int e = 0; e < 16; ++e) {
      int ka = ((e >> 3) << 4) + (half << 3) + (e & 7);   // A: K = (e/8)*16 + half*8 + e%8
      ((__bf16*)&af)[e] = f2bf(As[abuf][wrow * 16 + l16][ka]);
      int kb = ((e >> 3) << 4) + (half << 3) + (e & 7);   // B: K = (e/8)*16 + half*8 + e%8
      float bv = (BTRANS == 0) ? Bs[bbuf][nloc][kb] : Bs[bbuf][kb][nloc];
      ((__bf16*)&bf)[e] = f2bf(bv);
    }
    acc = __builtin_amdgcn_wmma_f32_16x16x32_bf16(false, af, false, bf,
                                                  (short)0, acc, false, false);
    __syncthreads();
  }
  // store: C/D layout: m = r + 8*half, n = lane&15
  const int gn = bn + wcol * 16 + l16;
  #pragma unroll
  for (int r = 0; r < 8; ++r) {
    int gm = bm + wrow * 16 + (half << 3) + r;
    if (gm < p.M && gn < p.N) {
      float v = acc[r];
      if (p.bias) v += p.bias[gn];
      if (p.beta != 0.0f) v += p.beta * p.C[(size_t)gm * p.N + gn];
      v *= p.out_scale;
      if (p.act == 1)      v = v > 0.0f ? v : 0.0f;
      else if (p.act == 2) v = v > 0.0f ? v : 0.01f * v;
      p.C[(size_t)gm * p.N + gn] = v;
    }
  }
}

// ---------------- helper kernels ----------------
__global__ void embed_nlc_k(const int* tok, const float* table, float* out, int BL, int D) {
  int i = blockIdx.x * blockDim.x + threadIdx.x;
  if (i >= BL * D) return;
  int bl = i / D, c = i - bl * D;
  out[i] = table[(size_t)tok[bl] * D + c];
}

__global__ void node_counts_k(const int* tok, float* cnt, int BN, int V, int L, float inv) {
  int i = blockIdx.x * blockDim.x + threadIdx.x;
  if (i >= BN * V) return;
  int bn = i / V, v = i - bn * V;
  const int* t = tok + (size_t)bn * L;
  int c = 0;
  for (int l = 0; l < L; ++l) c += (t[l] == v);
  cnt[i] = (float)c * inv;
}

__global__ void maxpool_nlc_k(const float* x, float* out, int Bn, int L, int C) {
  int i = blockIdx.x * blockDim.x + threadIdx.x;
  if (i >= Bn * C) return;
  int b = i / C, c = i - b * C;
  float m = -3.0e38f;
  for (int l = 0; l < L; ++l) m = fmaxf(m, x[((size_t)b * L + l) * C + c]);
  out[i] = m;
}

__global__ void seg_accum_k(const float* x, const int* edges, int E, int srow, int drow,
                            float* agg, float* cnt, int Nn, int F) {
  int b = blockIdx.x, e = blockIdx.y;
  const int* eb = edges + (size_t)b * 2 * E;
  int s = eb[srow * E + e];
  int d = eb[drow * E + e];
  const float* xs = x + ((size_t)b * Nn + s) * F;
  float* ag = agg + ((size_t)b * Nn + d) * F;
  for (int f = threadIdx.x; f < F; f += blockDim.x) atomicAdd(&ag[f], xs[f]);
  if (threadIdx.x == 0) atomicAdd(&cnt[(size_t)b * Nn + d], 1.0f);
}

__global__ void seg_norm_k(float* agg, const float* cnt, int BN, int F) {
  int i = blockIdx.x * blockDim.x + threadIdx.x;
  if (i >= BN * F) return;
  agg[i] /= fmaxf(cnt[i / F], 1.0f);
}

__global__ void build_pair_k(const float* dfeat, const float* pfeat,
                             const float* dx, const float* px, float* pair) {
  int i = blockIdx.x * blockDim.x + threadIdx.x;
  if (i >= BSZ * 1420) return;
  int b = i / 1420, j = i - b * 1420;
  float v;
  if      (j < 160)  v = dfeat[b * 160 + j];
  else if (j < 320)  v = pfeat[b * 160 + (j - 160)];
  else if (j < 420)  v = dx[((size_t)b * NDN) * LDL + (j - 320)];       // dx[b][0][:]
  else               v = px[((size_t)b * NPRN) * LPL + (j - 420)];      // px[b][0][:]
  pair[i] = v;
}

// ---------------- launch helpers ----------------
static inline unsigned magic_u32(unsigned d) {
  return (unsigned)(((1ull << 32) + d - 1) / d);   // exact for x*d < 2^32 (holds here)
}
static inline void gemm(hipStream_t st, const float* A, const float* Bw, const float* bias,
                        float* C, int M, int N, int K, int btrans, int act,
                        float beta, float oscale,
                        int amode = 0, int L = 0, int Cch = 0, int KW = 0, int Lout = 0) {
  GemmP p; p.A = A; p.Bw = Bw; p.bias = bias; p.C = C;
  p.M = M; p.N = N; p.K = K; p.L = L; p.Cch = Cch; p.KW = KW; p.Lout = Lout;
  p.mLout = Lout > 0 ? magic_u32((unsigned)Lout) : 0u;
  p.mKW   = KW   > 0 ? magic_u32((unsigned)KW)   : 0u;
  p.act = act; p.beta = beta; p.out_scale = oscale;
  dim3 g((N + 31) / 32, (M + 63) / 64);
  if (amode == 0 && btrans == 0)      gemm_bf16_wmma<0, 0><<<g, dim3(256), 0, st>>>(p);
  else if (amode == 0)                gemm_bf16_wmma<0, 1><<<g, dim3(256), 0, st>>>(p);
  else                                gemm_bf16_wmma<1, 0><<<g, dim3(256), 0, st>>>(p);
}
static inline int cdiv(int a, int b) { return (a + b - 1) / b; }

// ---------------- driver ----------------
extern "C" void kernel_launch(void* const* d_in, const int* in_sizes, int n_in,
                              void* d_out, int out_size, void* d_ws, size_t ws_size,
                              hipStream_t stream) {
  (void)in_sizes; (void)n_in; (void)out_size; (void)ws_size;
  // inputs
  const int*   drug_tok   = (const int*)  d_in[0];
  const int*   prot_tok   = (const int*)  d_in[1];
  const int*   dnode_tok  = (const int*)  d_in[2];
  const int*   pnode_tok  = (const int*)  d_in[3];
  const int*   ddi        = (const int*)  d_in[4];
  const int*   ppi        = (const int*)  d_in[5];
  const int*   dpi        = (const int*)  d_in[6];
  const float* d_embed    = (const float*)d_in[7];
  const float* p_embed    = (const float*)d_in[8];
  const float* dW1 = (const float*)d_in[9],  *db1 = (const float*)d_in[10];
  const float* dW2 = (const float*)d_in[11], *db2 = (const float*)d_in[12];
  const float* dW3 = (const float*)d_in[13], *db3 = (const float*)d_in[14];
  const float* pW1 = (const float*)d_in[15], *pb1 = (const float*)d_in[16];
  const float* pW2 = (const float*)d_in[17], *pb2 = (const float*)d_in[18];
  const float* pW3 = (const float*)d_in[19], *pb3 = (const float*)d_in[20];
  const float* dnode_tab  = (const float*)d_in[21];
  const float* pnode_tab  = (const float*)d_in[22];
  const float* ddi_Wl = (const float*)d_in[23], *ddi_bl = (const float*)d_in[24], *ddi_Wr = (const float*)d_in[25];
  const float* ppi_Wl = (const float*)d_in[26], *ppi_bl = (const float*)d_in[27], *ppi_Wr = (const float*)d_in[28];
  const float* dpi_Wl = (const float*)d_in[29], *dpi_bl = (const float*)d_in[30], *dpi_Wr = (const float*)d_in[31];
  const float* pdi_Wl = (const float*)d_in[32], *pdi_bl = (const float*)d_in[33], *pdi_Wr = (const float*)d_in[34];
  const float* fc1_W = (const float*)d_in[35], *fc1_b = (const float*)d_in[36];
  const float* fc2_W = (const float*)d_in[37], *fc2_b = (const float*)d_in[38];
  const float* fc3_W = (const float*)d_in[39], *fc3_b = (const float*)d_in[40];
  const float* out_W = (const float*)d_in[41], *out_b = (const float*)d_in[42];
  float* out = (float*)d_out;

  // workspace arena (f32, 256B aligned slices)
  char* w = (char*)d_ws;
  auto alloc_f = [&](size_t n) {
    float* p = (float*)w;
    w += ((n * sizeof(float) + 255) & ~(size_t)255);
    return p;
  };
  float* d_emb   = alloc_f((size_t)BSZ * LDL * DIMC);        // (B,100,64) NLC
  float* p_emb   = alloc_f((size_t)BSZ * LPL * DIMC);        // (B,1000,64) NLC
  float* d_act1  = alloc_f((size_t)BSZ * 97  * 40);
  float* d_act2  = alloc_f((size_t)BSZ * 92  * 80);
  float* d_act3  = alloc_f((size_t)BSZ * 85  * 160);
  float* p_act1  = alloc_f((size_t)BSZ * 997 * 40);
  float* p_act2  = alloc_f((size_t)BSZ * 990 * 80);
  float* p_act3  = alloc_f((size_t)BSZ * 979 * 160);
  float* dfeat   = alloc_f((size_t)BSZ * 160);
  float* pfeat   = alloc_f((size_t)BSZ * 160);
  float* d_cnts  = alloc_f((size_t)BSZ * NDN  * VDV);
  float* p_cnts  = alloc_f((size_t)BSZ * NPRN * VPV);
  float* dx_a    = alloc_f((size_t)BSZ * NDN  * LDL);
  float* dx_b    = alloc_f((size_t)BSZ * NDN  * LDL);
  float* px_a    = alloc_f((size_t)BSZ * NPRN * LPL);
  float* px_b    = alloc_f((size_t)BSZ * NPRN * LPL);
  char*  agg_begin = w;
  float* agg_dd  = alloc_f((size_t)BSZ * NDN  * LDL);        // drug<-drug sums
  float* agg_dp  = alloc_f((size_t)BSZ * NDN  * LPL);        // drug<-protein sums
  float* agg_pp  = alloc_f((size_t)BSZ * NPRN * LPL);        // protein<-protein sums
  float* agg_pd  = alloc_f((size_t)BSZ * NPRN * LDL);        // protein<-drug sums
  float* cnt_dd  = alloc_f((size_t)BSZ * NDN);
  float* cnt_dp  = alloc_f((size_t)BSZ * NDN);
  float* cnt_pp  = alloc_f((size_t)BSZ * NPRN);
  float* cnt_pd  = alloc_f((size_t)BSZ * NPRN);
  char*  agg_end = w;
  float* pair    = alloc_f((size_t)BSZ * 1420);
  float* h1      = alloc_f((size_t)BSZ * 1024);
  float* h2      = alloc_f((size_t)BSZ * 1024);
  float* h3      = alloc_f((size_t)BSZ * 512);

  const int TB = 256;

  // ---- 1) CNN paths (embed -> 3x implicit-im2col WMMA conv+ReLU -> maxpool) ----
  embed_nlc_k<<<cdiv(BSZ * LDL * DIMC, TB), TB, 0, stream>>>(drug_tok, d_embed, d_emb, BSZ * LDL, DIMC);
  embed_nlc_k<<<cdiv(BSZ * LPL * DIMC, TB), TB, 0, stream>>>(prot_tok, p_embed, p_emb, BSZ * LPL, DIMC);

  // drug: (B,100,64)->(B,97,40)->(B,92,80)->(B,85,160)
  gemm(stream, d_emb,  dW1, db1, d_act1, BSZ * 97, 40,  64 * 4,  0, 1, 0.f, 1.f, 1, 100, 64, 4, 97);
  gemm(stream, d_act1, dW2, db2, d_act2, BSZ * 92, 80,  40 * 6,  0, 1, 0.f, 1.f, 1, 97,  40, 6, 92);
  gemm(stream, d_act2, dW3, db3, d_act3, BSZ * 85, 160, 80 * 8,  0, 1, 0.f, 1.f, 1, 92,  80, 8, 85);
  maxpool_nlc_k<<<cdiv(BSZ * 160, TB), TB, 0, stream>>>(d_act3, dfeat, BSZ, 85, 160);

  // protein: (B,1000,64)->(B,997,40)->(B,990,80)->(B,979,160)
  gemm(stream, p_emb,  pW1, pb1, p_act1, BSZ * 997, 40,  64 * 4,   0, 1, 0.f, 1.f, 1, 1000, 64, 4,  997);
  gemm(stream, p_act1, pW2, pb2, p_act2, BSZ * 990, 80,  40 * 8,   0, 1, 0.f, 1.f, 1, 997,  40, 8,  990);
  gemm(stream, p_act2, pW3, pb3, p_act3, BSZ * 979, 160, 80 * 12,  0, 1, 0.f, 1.f, 1, 990,  80, 12, 979);
  maxpool_nlc_k<<<cdiv(BSZ * 160, TB), TB, 0, stream>>>(p_act3, pfeat, BSZ, 979, 160);

  // ---- 2) node features: counts/L @ table  (b_trans=1: table is (V, F)) ----
  node_counts_k<<<cdiv(BSZ * NDN * VDV, TB), TB, 0, stream>>>(dnode_tok, d_cnts, BSZ * NDN, VDV, LDL, 1.0f / LDL);
  node_counts_k<<<cdiv(BSZ * NPRN * VPV, TB), TB, 0, stream>>>(pnode_tok, p_cnts, BSZ * NPRN, VPV, LPL, 1.0f / LPL);
  gemm(stream, d_cnts, dnode_tab, nullptr, dx_a, BSZ * NDN,  LDL, VDV, 1, 0, 0.f, 1.f);
  gemm(stream, p_cnts, pnode_tab, nullptr, px_a, BSZ * NPRN, LPL, VPV, 1, 0, 0.f, 1.f);

  // ---- 3) hetero-SAGE GNN, 2 layers ----
  float* dx_cur = dx_a; float* dx_nxt = dx_b;
  float* px_cur = px_a; float* px_nxt = px_b;
  for (int layer = 0; layer < 2; ++layer) {
    (void)hipMemsetAsync(agg_begin, 0, (size_t)(agg_end - agg_begin), stream);
    // segment sums (pdi = dpi reversed: src=row1, dst=row0)
    seg_accum_k<<<dim3(BSZ, EDD), 128, 0, stream>>>(dx_cur, ddi, EDD, 0, 1, agg_dd, cnt_dd, NDN,  LDL);
    seg_accum_k<<<dim3(BSZ, EDP), 128, 0, stream>>>(px_cur, dpi, EDP, 1, 0, agg_dp, cnt_dp, NDN,  LPL);
    seg_accum_k<<<dim3(BSZ, EPP), 128, 0, stream>>>(px_cur, ppi, EPP, 0, 1, agg_pp, cnt_pp, NPRN, LPL);
    seg_accum_k<<<dim3(BSZ, EDP), 128, 0, stream>>>(dx_cur, dpi, EDP, 0, 1, agg_pd, cnt_pd, NPRN, LDL);
    seg_norm_k<<<cdiv(BSZ * NDN  * LDL, TB), TB, 0, stream>>>(agg_dd, cnt_dd, BSZ * NDN,  LDL);
    seg_norm_k<<<cdiv(BSZ * NDN  * LPL, TB), TB, 0, stream>>>(agg_dp, cnt_dp, BSZ * NDN,  LPL);
    seg_norm_k<<<cdiv(BSZ * NPRN * LPL, TB), TB, 0, stream>>>(agg_pp, cnt_pp, BSZ * NPRN, LPL);
    seg_norm_k<<<cdiv(BSZ * NPRN * LDL, TB), TB, 0, stream>>>(agg_pd, cnt_pd, BSZ * NPRN, LDL);
    // d_new = 0.5*(agg_dd@ddi_Wl^T + ddi_bl + dx@ddi_Wr^T + agg_dp@pdi_Wl^T + pdi_bl + dx@pdi_Wr^T)
    const int MD = BSZ * NDN;
    gemm(stream, agg_dd, ddi_Wl, ddi_bl, dx_nxt, MD, LDL, LDL, 0, 0, 0.f, 1.f);
    gemm(stream, dx_cur, ddi_Wr, nullptr, dx_nxt, MD, LDL, LDL, 0, 0, 1.f, 1.f);
    gemm(stream, agg_dp, pdi_Wl, pdi_bl, dx_nxt, MD, LDL, LPL, 0, 0, 1.f, 1.f);
    gemm(stream, dx_cur, pdi_Wr, nullptr, dx_nxt, MD, LDL, LDL, 0, 0, 1.f, 0.5f);
    // p_new = 0.5*(agg_pp@ppi_Wl^T + ppi_bl + px@ppi_Wr^T + agg_pd@dpi_Wl^T + dpi_bl + px@dpi_Wr^T)
    const int MP = BSZ * NPRN;
    gemm(stream, agg_pp, ppi_Wl, ppi_bl, px_nxt, MP, LPL, LPL, 0, 0, 0.f, 1.f);
    gemm(stream, px_cur, ppi_Wr, nullptr, px_nxt, MP, LPL, LPL, 0, 0, 1.f, 1.f);
    gemm(stream, agg_pd, dpi_Wl, dpi_bl, px_nxt, MP, LPL, LDL, 0, 0, 1.f, 1.f);
    gemm(stream, px_cur, dpi_Wr, nullptr, px_nxt, MP, LPL, LPL, 0, 0, 1.f, 0.5f);
    // swap
    float* t;
    t = dx_cur; dx_cur = dx_nxt; dx_nxt = t;
    t = px_cur; px_cur = px_nxt; px_nxt = t;
  }

  // ---- 4) head: concat -> 3x leaky FC -> linear out ----
  build_pair_k<<<cdiv(BSZ * 1420, TB), TB, 0, stream>>>(dfeat, pfeat, dx_cur, px_cur, pair);
  gemm(stream, pair, fc1_W, fc1_b, h1,  BSZ, 1024, 1420, 0, 2, 0.f, 1.f);
  gemm(stream, h1,   fc2_W, fc2_b, h2,  BSZ, 1024, 1024, 0, 2, 0.f, 1.f);
  gemm(stream, h2,   fc3_W, fc3_b, h3,  BSZ, 512,  1024, 0, 2, 0.f, 1.f);
  gemm(stream, h3,   out_W, out_b, out, BSZ, 2,    512,  0, 0, 0.f, 1.f);
}
